// GAT_50L_uw_54485955117459
// MI455X (gfx1250) — matrix operand order repaired and checked by reading, not compiled, not run on hardware
//
#include <hip/hip_runtime.h>
#include <hip/hip_bf16.h>
#include <stdint.h>

// ---------------------------------------------------------------------------
// GATv2 x50 for MI455X (gfx1250): bf16 WMMA GEMMs (V_WMMA_F32_16X16X32_BF16)
// with A-fragment reuse across N-tiles and fused L/R projections,
// L2-resident edge softmax via atomics, fused bias/ReLU/BatchNorm.
// ---------------------------------------------------------------------------

typedef __attribute__((ext_vector_type(16))) __bf16 v16bf;
typedef __attribute__((ext_vector_type(8)))  __bf16 v8bf;
typedef __attribute__((ext_vector_type(8)))  float  v8f;

#define IN_F   256
#define HID    128
#define HEADS  4
#define CCH    32
#define LMID   48

// ---- helpers ---------------------------------------------------------------

__device__ __forceinline__ __bf16 f2bf(float f) {
  unsigned u = __float_as_uint(f);
  unsigned r = u + 0x7FFFu + ((u >> 16) & 1u);   // round-to-nearest-even
  unsigned short s = (unsigned short)(r >> 16);
  __bf16 b;
  __builtin_memcpy(&b, &s, sizeof(b));
  return b;
}

// monotonic float<->uint mapping so atomicMax(u32) == float max
__device__ __forceinline__ unsigned fenc(float f) {
  unsigned u = __float_as_uint(f);
  return (u & 0x80000000u) ? ~u : (u | 0x80000000u);
}
__device__ __forceinline__ float fdec(unsigned k) {
  unsigned u = (k & 0x80000000u) ? (k & 0x7FFFFFFFu) : ~k;
  return __uint_as_float(u);
}

__device__ __forceinline__ v16bf combine16(v8bf lo, v8bf hi) {
  v16bf r;
#pragma unroll
  for (int i = 0; i < 8; ++i) { r[i] = lo[i]; r[i + 8] = hi[i]; }
  return r;
}

// ---- utility kernels -------------------------------------------------------

__global__ void k_fill0(float* __restrict__ p, long n) {
  long i = (long)blockIdx.x * blockDim.x + threadIdx.x;
  long st = (long)gridDim.x * blockDim.x;
  for (; i < n; i += st) p[i] = 0.0f;
}

__global__ void k_f2bf(const float* __restrict__ x, __bf16* __restrict__ y, long n) {
  long i = (long)blockIdx.x * blockDim.x + threadIdx.x;
  long st = (long)gridDim.x * blockDim.x;
  for (; i < n; i += st) y[i] = f2bf(x[i]);
}

// W [K,Nout] f32 row-major  ->  Wt [Nout,K] bf16 row-major (transpose+convert)
__global__ void k_wt(const float* __restrict__ W, __bf16* __restrict__ Wt,
                     int K, int Nout) {
  int idx = blockIdx.x * blockDim.x + threadIdx.x;
  if (idx >= K * Nout) return;
  int n = idx / K, k = idx - n * K;
  Wt[idx] = f2bf(W[(size_t)k * Nout + n]);
}

// ---- WMMA dual GEMM: CL = A@Wl, CR = A@Wr ----------------------------------
// A  : [M,K] bf16 row-major (activations)
// Bt : [Nout,K] bf16 row-major (pre-transposed weights -> per-lane contiguous)
// Each wave owns a 16-row x (NT*16)-col strip of BOTH CL and CR, reusing the
// A fragment across 2*NT WMMAs per K-step (v_wmma_f32_16x16x32_bf16).
template <int NT>
__global__ void k_gemm_dual(const __bf16* __restrict__ A,
                            const __bf16* __restrict__ BtL,
                            const __bf16* __restrict__ BtR,
                            float* __restrict__ CL,
                            float* __restrict__ CR,
                            int M, int K, int Nout) {
  const int lane = threadIdx.x & 31;
  const int wave = threadIdx.x >> 5;
  const int tilesM = M >> 4, tilesN = Nout >> 4;
  const int strips = tilesN / NT;
  int sid = blockIdx.x * (blockDim.x >> 5) + wave;
  if (sid >= tilesM * strips) return;
  const int tm = sid / strips, sn = sid - tm * strips;
  const int half = lane >> 4, l15 = lane & 15;

  // A lane layout: lanes 0-15 row M=l15, K = 0..15;
  // lanes 16-31 same row, K = 16..31
  const __bf16* ap = A + (size_t)(tm * 16 + l15) * K + half * 16;

  v8f accL[NT] = {};
  v8f accR[NT] = {};
  for (int ks = 0; ks < K; ks += 32) {
    v8bf a0 = *(const v8bf*)(ap + ks);
    v8bf a1 = *(const v8bf*)(ap + ks + 8);
    v16bf av = combine16(a0, a1);
#pragma unroll
    for (int u = 0; u < NT; ++u) {
      // B lane layout: lane holds column N=l15; lanes 0-15 K=0..15, 16-31 K=16..31
      const size_t boff = (size_t)((sn * NT + u) * 16 + l15) * K + half * 16 + ks;
      const __bf16* bl = BtL + boff;
      const __bf16* br = BtR + boff;
      v16bf bv = combine16(*(const v8bf*)bl, *(const v8bf*)(bl + 8));
      accL[u] = __builtin_amdgcn_wmma_f32_16x16x32_bf16(
          false, av, false, bv, (short)0, accL[u], false, false);
      v16bf cv = combine16(*(const v8bf*)br, *(const v8bf*)(br + 8));
      accR[u] = __builtin_amdgcn_wmma_f32_16x16x32_bf16(
          false, av, false, cv, (short)0, accR[u], false, false);
    }
  }
  // C/D layout: VGPR v -> row tm*16 + v + half*8, col tn*16 + l15
#pragma unroll
  for (int u = 0; u < NT; ++u) {
    const size_t coff = (size_t)(tm * 16 + half * 8) * Nout +
                        (sn * NT + u) * 16 + l15;
    float* cpl = CL + coff;
    float* cpr = CR + coff;
#pragma unroll
    for (int v = 0; v < 8; ++v) {
      cpl[(size_t)v * Nout] = accL[u][v];
      cpr[(size_t)v * Nout] = accR[u][v];
    }
  }
}

// ---- edge kernels ----------------------------------------------------------

__device__ __forceinline__ void edge_sd(const int* __restrict__ ei, int E,
                                        int e, int& s, int& d) {
  if (e < E) { s = ei[e]; d = ei[E + e]; }
  else       { s = d = e - E; }          // self-loop
}

// logits + segment-max (per dst node, per head)
__global__ void k_edge_logit(const float* __restrict__ xl,
                             const float* __restrict__ xr,
                             const float* __restrict__ att,
                             const int* __restrict__ ei,
                             int E, int N, int Hh, int Cc,
                             float* __restrict__ logit,
                             unsigned* __restrict__ menc) {
  long idx = (long)blockIdx.x * blockDim.x + threadIdx.x;
  long tot = (long)(E + N) * Hh;
  if (idx >= tot) return;
  int e = (int)(idx / Hh), h = (int)(idx - (long)e * Hh);
  int s, d; edge_sd(ei, E, e, s, d);
  const int D = Hh * Cc;
  const float* pl = xl + (size_t)s * D + h * Cc;
  const float* pr = xr + (size_t)d * D + h * Cc;
  const float* pa = att + h * Cc;
  float acc = 0.f;
#pragma unroll 8
  for (int c = 0; c < Cc; ++c) {
    float v = pl[c] + pr[c];
    v = v > 0.f ? v : 0.2f * v;          // leaky_relu(0.2)
    acc += v * pa[c];
  }
  logit[idx] = acc;
  atomicMax(&menc[(size_t)d * Hh + h], fenc(acc));
}

// p = exp(logit - max) ; segment-sum z
__global__ void k_edge_exp(const float* __restrict__ logit,
                           const unsigned* __restrict__ menc,
                           float* __restrict__ z,
                           const int* __restrict__ ei,
                           int E, int N, int Hh,
                           float* __restrict__ pexp) {
  long idx = (long)blockIdx.x * blockDim.x + threadIdx.x;
  long tot = (long)(E + N) * Hh;
  if (idx >= tot) return;
  int e = (int)(idx / Hh), h = (int)(idx - (long)e * Hh);
  int s, d; edge_sd(ei, E, e, s, d);
  float mm = fdec(menc[(size_t)d * Hh + h]);
  float pv = __expf(logit[idx] - mm);
  pexp[idx] = pv;
  atomicAdd(&z[(size_t)d * Hh + h], pv);
}

// out[dst] += alpha * xl[src]   (thread per edge-feature)
__global__ void k_edge_aggr(const float* __restrict__ xl,
                            const float* __restrict__ pexp,
                            const float* __restrict__ z,
                            const int* __restrict__ ei,
                            int E, int N, int Hh, int Cc,
                            float* __restrict__ agg) {
  const int D = Hh * Cc;
  long idx = (long)blockIdx.x * blockDim.x + threadIdx.x;
  long tot = (long)(E + N) * D;
  if (idx >= tot) return;
  int e = (int)(idx / D), j = (int)(idx - (long)e * D);
  int h = j / Cc;
  int s, d; edge_sd(ei, E, e, s, d);
  float alpha = pexp[(size_t)e * Hh + h] / (z[(size_t)d * Hh + h] + 1e-16f);
  atomicAdd(&agg[(size_t)d * D + j], alpha * xl[(size_t)s * D + j]);
}

// ---- fused bias/ReLU/BatchNorm --------------------------------------------

// column sums of relu(agg+bias): blockDim.x == D, grid-strided over rows
__global__ void k_bn_stats(const float* __restrict__ agg,
                           const float* __restrict__ bias,
                           float* __restrict__ stats,  // [2*D] pre-zeroed
                           int N, int D, int relu) {
  int t = threadIdx.x;
  float s = 0.f, s2 = 0.f;
  for (int r = blockIdx.x; r < N; r += gridDim.x) {
    float v = agg[(size_t)r * D + t] + bias[t];
    if (relu) v = fmaxf(v, 0.f);
    s += v; s2 += v * v;
  }
  atomicAdd(&stats[t], s);
  atomicAdd(&stats[D + t], s2);
}

__global__ void k_bn_norm(const float* __restrict__ agg,
                          const float* __restrict__ bias,
                          const float* __restrict__ stats,
                          const float* __restrict__ gamma,
                          const float* __restrict__ beta,
                          float* __restrict__ hf, __bf16* __restrict__ hb,
                          int N, int D, int relu) {
  long idx = (long)blockIdx.x * blockDim.x + threadIdx.x;
  long tot = (long)N * D;
  if (idx >= tot) return;
  int j = (int)(idx % D);
  float invN = 1.0f / (float)N;
  float mu  = stats[j] * invN;
  float var = stats[D + j] * invN - mu * mu;
  float v = agg[idx] + bias[j];
  if (relu) v = fmaxf(v, 0.f);
  float y = gamma[j] * (v - mu) * rsqrtf(var + 1e-5f) + beta[j];
  hf[idx] = y;
  hb[idx] = f2bf(y);
}

// ---- output linear [N,C] @ [C,O] + b --------------------------------------

__global__ void k_lin(const float* __restrict__ h, const float* __restrict__ w,
                      const float* __restrict__ b, float* __restrict__ out,
                      int N, int C, int O) {
  int i = blockIdx.x * blockDim.x + threadIdx.x;
  if (i >= N) return;
  for (int o = 0; o < O; ++o) {
    float acc = b[o];
    for (int c = 0; c < C; ++c) acc += h[(size_t)i * C + c] * w[c * O + o];
    out[(size_t)i * O + o] = acc;
  }
}

// ---------------------------------------------------------------------------

extern "C" void kernel_launch(void* const* d_in, const int* in_sizes, int n_in,
                              void* d_out, int out_size, void* d_ws, size_t ws_size,
                              hipStream_t stream) {
  const float* x    = (const float*)d_in[0];
  const int*   ei   = (const int*)  d_in[1];
  const float* W0l  = (const float*)d_in[2];
  const float* W0r  = (const float*)d_in[3];
  const float* att0 = (const float*)d_in[4];
  const float* b0   = (const float*)d_in[5];
  const float* Wml  = (const float*)d_in[6];
  const float* Wmr  = (const float*)d_in[7];
  const float* attm = (const float*)d_in[8];
  const float* bm   = (const float*)d_in[9];
  const float* Wfl  = (const float*)d_in[10];
  const float* Wfr  = (const float*)d_in[11];
  const float* attf = (const float*)d_in[12];
  const float* bfb  = (const float*)d_in[13];
  const float* bng  = (const float*)d_in[14];
  const float* bnb  = (const float*)d_in[15];
  const float* bnfg = (const float*)d_in[16];
  const float* bnfb = (const float*)d_in[17];
  const float* linw = (const float*)d_in[18];
  const float* linb = (const float*)d_in[19];
  (void)n_in; (void)out_size; (void)ws_size;

  const int N  = in_sizes[0] / IN_F;
  const int E  = in_sizes[1] / 2;
  const int ET = E + N;

  // ---- workspace layout (256B aligned slices) ----
  char* wp = (char*)d_ws;
  auto alloc = [&](size_t bytes) -> char* {
    char* p = wp;
    wp += (bytes + 255) & ~(size_t)255;
    return p;
  };
  __bf16* xbf  = (__bf16*)alloc((size_t)N * IN_F * sizeof(__bf16));
  __bf16* hbf  = (__bf16*)alloc((size_t)N * HID * sizeof(__bf16));
  float*  hfl  = (float*) alloc((size_t)N * HID * sizeof(float));
  float*  xl   = (float*) alloc((size_t)N * HID * sizeof(float));
  float*  xr   = (float*) alloc((size_t)N * HID * sizeof(float));
  float*  logit= (float*) alloc((size_t)ET * HEADS * sizeof(float));
  float*  pexp = (float*) alloc((size_t)ET * HEADS * sizeof(float));
  // contiguous zero-block: m_enc | z | agg | stats
  float*  zblk = (float*) alloc(((size_t)N*HEADS*2 + (size_t)N*HID + 2*HID) * sizeof(float));
  float*  menc = zblk;
  float*  zsum = zblk + (size_t)N * HEADS;
  float*  agg  = zblk + (size_t)N * HEADS * 2;
  float*  stats= agg  + (size_t)N * HID;
  const long zTotal = (long)N * HEADS * 2 + (long)N * HID + 2 * HID;
  __bf16* W0tl = (__bf16*)alloc((size_t)IN_F * HID * sizeof(__bf16));
  __bf16* W0tr = (__bf16*)alloc((size_t)IN_F * HID * sizeof(__bf16));
  __bf16* Wmtl = (__bf16*)alloc((size_t)LMID * HID * HID * sizeof(__bf16));
  __bf16* Wmtr = (__bf16*)alloc((size_t)LMID * HID * HID * sizeof(__bf16));
  __bf16* Wftl = (__bf16*)alloc((size_t)HID * CCH * sizeof(__bf16));
  __bf16* Wftr = (__bf16*)alloc((size_t)HID * CCH * sizeof(__bf16));

  const int T = 256;

  // ---- one-time (per call) weight/feature conversion ----
  k_f2bf<<<512, T, 0, stream>>>(x, xbf, (long)N * IN_F);
  {
    int n0 = IN_F * HID;
    k_wt<<<(n0 + T - 1) / T, T, 0, stream>>>(W0l, W0tl, IN_F, HID);
    k_wt<<<(n0 + T - 1) / T, T, 0, stream>>>(W0r, W0tr, IN_F, HID);
    int nm = HID * HID;
    for (int l = 0; l < LMID; ++l) {
      k_wt<<<(nm + T - 1) / T, T, 0, stream>>>(Wml + (size_t)l * nm,
                                               Wmtl + (size_t)l * nm, HID, HID);
      k_wt<<<(nm + T - 1) / T, T, 0, stream>>>(Wmr + (size_t)l * nm,
                                               Wmtr + (size_t)l * nm, HID, HID);
    }
    int nf = HID * CCH;
    k_wt<<<(nf + T - 1) / T, T, 0, stream>>>(Wfl, Wftl, HID, CCH);
    k_wt<<<(nf + T - 1) / T, T, 0, stream>>>(Wfr, Wftr, HID, CCH);
  }

  // ---- one GATv2 layer ----
  auto run_layer = [&](const __bf16* Abf, int K, const __bf16* Btl,
                       const __bf16* Btr, const float* att, const float* bias,
                       const float* gamma, const float* beta, int Hh, int Cc,
                       int relu) {
    const int D = Hh * Cc;
    k_fill0<<<512, T, 0, stream>>>(zblk, zTotal);

    // WMMA strip GEMM: NT=2 N-tiles per wave, L/R fused (A reused 4x)
    int strips = (N / 16) * ((D / 16) / 2);
    k_gemm_dual<2><<<(strips + 7) / 8, 256, 0, stream>>>(Abf, Btl, Btr,
                                                         xl, xr, N, K, D);
    long nh = (long)ET * Hh;
    k_edge_logit<<<(int)((nh + T - 1) / T), T, 0, stream>>>(
        xl, xr, att, ei, E, N, Hh, Cc, logit, (unsigned*)menc);
    k_edge_exp<<<(int)((nh + T - 1) / T), T, 0, stream>>>(
        logit, (const unsigned*)menc, zsum, ei, E, N, Hh, pexp);
    long nd = (long)ET * D;
    k_edge_aggr<<<(int)((nd + T - 1) / T), T, 0, stream>>>(
        xl, pexp, zsum, ei, E, N, Hh, Cc, agg);
    k_bn_stats<<<128, D, 0, stream>>>(agg, bias, stats, N, D, relu);
    long nt = (long)N * D;
    k_bn_norm<<<(int)((nt + T - 1) / T), T, 0, stream>>>(
        agg, bias, stats, gamma, beta, hfl, hbf, N, D, relu);
  };

  // layer 0: IN_F -> HID, relu, bn[0]
  run_layer(xbf, IN_F, W0tl, W0tr, att0, b0, bng, bnb, HEADS, CCH, 1);
  // layers 1..48
  for (int l = 0; l < LMID; ++l) {
    run_layer(hbf, HID,
              Wmtl + (size_t)l * HID * HID, Wmtr + (size_t)l * HID * HID,
              attm + (size_t)l * HEADS * CCH, bm + (size_t)l * HID,
              bng + (size_t)(l + 1) * HID, bnb + (size_t)(l + 1) * HID,
              HEADS, CCH, 1);
  }
  // final GATv2: HID -> C, 1 head, no relu, bnf
  run_layer(hbf, HID, Wftl, Wftr, attf, bfb, bnfg, bnfb, 1, CCH, 0);

  // output linear C -> 2
  k_lin<<<(N + T - 1) / T, T, 0, stream>>>(hfl, linw, linb, (float*)d_out,
                                           N, CCH, 2);
}
